// SimpleMultiheadAttention_82179904241864
// MI455X (gfx1250) — matrix operand order repaired and checked by reading, not compiled, not run on hardware
//
#include <hip/hip_runtime.h>
#include <hip/hip_bf16.h>
#include <stdint.h>

#ifndef __has_builtin
#define __has_builtin(x) 0
#endif

// Flip to 0 if the async-to-LDS builtin/cast fails to compile on this toolchain.
#define TRY_ASYNC 1

#if TRY_ASYNC && __has_builtin(__builtin_amdgcn_global_load_async_to_lds_b128)
#define HAVE_ASYNC_LDS 1
#else
#define HAVE_ASYNC_LDS 0
#endif

// ---------------- types for WMMA fragments ----------------
typedef __attribute__((ext_vector_type(16))) __bf16     v16bf;
typedef __attribute__((ext_vector_type(8)))  float      v8f;
typedef __attribute__((ext_vector_type(8)))  unsigned   v8u;

// vector-of-4-int type matching the async-builtin's pointer parameters
typedef int vi4 __attribute__((vector_size(4 * sizeof(int))));

union Frag16 { v16bf bf; v8u u; };

// ---------------- problem constants ----------------
#define B_   2
#define L_   2048
#define D_   1024
#define H_   16
#define DH_  64
#define M_   (B_ * L_)     // 4096
#define N1_  (3 * D_)      // 3072
#define SCALE_LOG2E (0.125f * 1.4426950408889634f)  // DH^-0.5 * log2(e)

// ---------------- helpers ----------------
static __device__ inline unsigned short f2bf(float f) {
  unsigned u = __builtin_bit_cast(unsigned, f);
  u += 0x7FFFu + ((u >> 16) & 1u);   // round-to-nearest-even
  return (unsigned short)(u >> 16);
}

static __device__ inline v8f wmma_bf16(const Frag16& a, const Frag16& b, v8f c) {
  return __builtin_amdgcn_wmma_f32_16x16x32_bf16(false, a.bf, false, b.bf,
                                                 (short)0, c, false, false);
}

// A fragment 16x32 (MxK) bf16 from row-major panel (16 rows, `stride` elems/row).
static __device__ inline void load_a(Frag16& a, const unsigned short* panel,
                                     int stride, int lane) {
  const int row = lane & 15, hi = (lane >> 4) * 16;
#pragma unroll
  for (int r = 0; r < 8; ++r) {
    const int kk = 2 * r + hi;
    a.u[r] = *(const unsigned*)(panel + row * stride + kk);
  }
}

// B fragment 32x16 (KxN) bf16; memory holds 16 rows indexed by N, K contiguous.
static __device__ inline void load_b(Frag16& b, const unsigned short* panel,
                                     long stride, int lane) {
  const int row = lane & 15, hi = (lane >> 4) * 16;
#pragma unroll
  for (int r = 0; r < 8; ++r) {
    const int kk = 2 * r + hi;
    b.u[r] = *(const unsigned*)(panel + (long)row * stride + kk);
  }
}

// ---- async global->LDS staging (16B per lane), with synchronous fallback ----
static __device__ inline void stage_b128(unsigned short* lds_dst,
                                         const unsigned short* gsrc) {
#if HAVE_ASYNC_LDS
  __builtin_amdgcn_global_load_async_to_lds_b128(
      (__attribute__((address_space(1))) vi4*)gsrc,
      (__attribute__((address_space(3))) vi4*)lds_dst, 0, 0);
#else
  *(uint4*)lds_dst = *(const uint4*)gsrc;
#endif
}

static __device__ inline void wait_async_le(int pending) {
#if HAVE_ASYNC_LDS
#if __has_builtin(__builtin_amdgcn_s_wait_asynccnt)
  if (pending) __builtin_amdgcn_s_wait_asynccnt(1);
  else         __builtin_amdgcn_s_wait_asynccnt(0);
#else
  if (pending) { asm volatile("s_wait_asynccnt 1" ::: "memory"); }
  else         { asm volatile("s_wait_asynccnt 0" ::: "memory"); }
#endif
#endif
}

// ---------------- stage 0: fp32 -> bf16 converts ----------------
__global__ void cvt_bf16_kernel(const float* __restrict__ in,
                                unsigned short* __restrict__ out, int n) {
  int i = blockIdx.x * blockDim.x + threadIdx.x;
  if (i < n) out[i] = f2bf(in[i]);
}

__global__ void cvt_t_bf16_kernel(const float* __restrict__ in,
                                  unsigned short* __restrict__ out, int R, int C) {
  int i = blockIdx.x * blockDim.x + threadIdx.x;
  if (i < R * C) {
    int r = i / C, c = i - r * C;
    out[(long)c * R + r] = f2bf(in[i]);
  }
}

// ---- shared GEMM mainloop: 256 threads, 8 waves, tile M=128 x N=64, K=D_.
// B panel (64 rows x 32 k) staged in LDS (double buffered, async copy);
// each wave owns a unique 16-row A panel (direct global, b128 loads).
static __device__ inline void gemm_mainloop(const unsigned short* Apanel,   // wave's 16 rows
                                            const unsigned short* Btpanel,  // block's 64 N-rows
                                            unsigned short* ldsB,           // [2][64*32]
                                            int tid, int lane, v8f c[4]) {
  const int srow = tid >> 2, spart = (tid & 3) * 8;   // 16B per thread, 4 KB total
  stage_b128(ldsB + srow * 32 + spart, Btpanel + (long)srow * D_ + spart);
  int buf = 0;
  for (int k0 = 0; k0 < D_; k0 += 32) {
    if (k0 + 32 < D_) {
      stage_b128(ldsB + (buf ^ 1) * (64 * 32) + srow * 32 + spart,
                 Btpanel + (long)srow * D_ + (k0 + 32) + spart);
      wait_async_le(1);
    } else {
      wait_async_le(0);
    }
    __syncthreads();
    Frag16 a;
    load_a(a, Apanel + k0, D_, lane);
    __builtin_prefetch(Apanel + k0 + 64, 0, 1);
    const unsigned short* bp = ldsB + buf * (64 * 32);
    {
      const int row = lane & 15, hi = (lane >> 4) * 16;
#pragma unroll
      for (int f = 0; f < 4; ++f) {
        Frag16 b;
#pragma unroll
        for (int r = 0; r < 8; ++r)
          b.u[r] = *(const unsigned*)(bp + (f * 16 + row) * 32 + 2 * r + hi);
        c[f] = wmma_bf16(a, b, c[f]);
      }
    }
    __syncthreads();   // protect buffer from next-next stage
    buf ^= 1;
  }
}

// ---------------- stage 1: QKV GEMM + scatter to per-head Q/K/V^T ----------------
__global__ __launch_bounds__(256) void gemm_qkv_kernel(
    const unsigned short* __restrict__ A,    // M_ x D_ bf16 (x)
    const unsigned short* __restrict__ Bt,   // N1_ x D_ bf16 (w_qkv^T)
    const float* __restrict__ bias,          // N1_
    unsigned short* __restrict__ Qd,         // (B*H, L, DH)
    unsigned short* __restrict__ Kd,         // (B*H, L, DH)
    unsigned short* __restrict__ Vt) {       // (B*H, DH, L)
  __shared__ unsigned short ldsB[2][64 * 32];
  const int tid = threadIdx.x, lane = tid & 31, wave = tid >> 5;
  const int n0 = blockIdx.x * 64;
  const int m0 = blockIdx.y * 128 + wave * 16;

  v8f c[4];
#pragma unroll
  for (int f = 0; f < 4; ++f)
#pragma unroll
    for (int i = 0; i < 8; ++i) c[f][i] = 0.f;

  gemm_mainloop(A + (long)m0 * D_, Bt + (long)n0 * D_, &ldsB[0][0], tid, lane, c);

  const int col = lane & 15, rowhi = (lane >> 4) * 8;
#pragma unroll
  for (int f = 0; f < 4; ++f) {
    const int n = n0 + f * 16 + col;
    const float bv = bias[n];
    const int sect = n >> 10, rem = n & 1023;
    const int h = rem >> 6, dh = rem & 63;
#pragma unroll
    for (int r = 0; r < 8; ++r) {
      const int m = m0 + r + rowhi;
      const int bi = m >> 11, l = m & (L_ - 1);
      const unsigned short hv = f2bf(c[f][r] + bv);
      const long bh = (long)bi * H_ + h;
      if (sect == 0)      Qd[(bh * L_ + l) * DH_ + dh] = hv;
      else if (sect == 1) Kd[(bh * L_ + l) * DH_ + dh] = hv;
      else                Vt[(bh * DH_ + dh) * L_ + l] = hv;
    }
  }
}

// ---------------- stage 2: flash attention, one wave = one 16-query tile ----------------
__global__ __launch_bounds__(128) void attn_kernel(
    const unsigned short* __restrict__ Qd,
    const unsigned short* __restrict__ Kd,
    const unsigned short* __restrict__ Vt,
    const unsigned char* __restrict__ mask,  // B_ x L_ (bool)
    unsigned short* __restrict__ Ctx) {      // M_ x D_ bf16
  __shared__ unsigned short lds_p[4][16 * 32];   // per-wave P staging (C->A relayout)

  const int lane = threadIdx.x & 31;
  const int wave = threadIdx.x >> 5;
  const int tile = blockIdx.x * 4 + wave;        // B*H*L/16 tiles total
  const int qt = tile & (L_ / 16 - 1);
  const int bh = tile / (L_ / 16);
  const int b = bh / H_, h = bh - b * H_;

  const unsigned short* qpan = Qd + ((long)bh * L_ + qt * 16) * DH_;
  Frag16 aq0, aq1;
  load_a(aq0, qpan, DH_, lane);        // dh 0..31
  load_a(aq1, qpan + 32, DH_, lane);   // dh 32..63

  v8f o[4];
#pragma unroll
  for (int f = 0; f < 4; ++f)
#pragma unroll
    for (int i = 0; i < 8; ++i) o[f][i] = 0.f;
  float mrow[8], lrow[8];
#pragma unroll
  for (int r = 0; r < 8; ++r) { mrow[r] = -INFINITY; lrow[r] = 0.f; }

  const unsigned char* mptr = mask + (long)b * L_;
  unsigned short* ldsp = lds_p[wave];
  const unsigned short* vpan = Vt + (long)bh * DH_ * L_;

  for (int kb = 0; kb < L_; kb += 32) {
    // ---- mask first: skip fully-masked 32-key blocks (wave-uniform branch)
    const int c0 = kb + (lane & 15);
    const bool msk0 = mptr[c0] != 0;
    const bool msk1 = mptr[c0 + 16] != 0;
    const unsigned allm = (unsigned)__ballot(msk0 && msk1);
    if (allm == 0xFFFFFFFFu) continue;   // identity update: p=0, alpha=1

    // ---- scores: S(16q x 32k) as two C tiles, K-dim = DH = 2 WMMAs each
    const unsigned short* kpan0 = Kd + ((long)bh * L_ + kb) * DH_;
    const unsigned short* kpan1 = kpan0 + 16 * DH_;
    v8f s0, s1;
#pragma unroll
    for (int i = 0; i < 8; ++i) { s0[i] = 0.f; s1[i] = 0.f; }
    Frag16 bk;
    load_b(bk, kpan0,      DH_, lane); s0 = wmma_bf16(aq0, bk, s0);
    load_b(bk, kpan0 + 32, DH_, lane); s0 = wmma_bf16(aq1, bk, s0);
    load_b(bk, kpan1,      DH_, lane); s1 = wmma_bf16(aq0, bk, s1);
    load_b(bk, kpan1 + 32, DH_, lane); s1 = wmma_bf16(aq1, bk, s1);

    // ---- scale + mask (log2 domain)
    float t0[8], t1[8];
#pragma unroll
    for (int r = 0; r < 8; ++r) {
      t0[r] = msk0 ? -INFINITY : s0[r] * SCALE_LOG2E;
      t1[r] = msk1 ? -INFINITY : s1[r] * SCALE_LOG2E;
    }

    // ---- online softmax: row reductions over 16-lane half-groups
    float mnew[8], alpha[8];
#pragma unroll
    for (int r = 0; r < 8; ++r) {
      float v = fmaxf(t0[r], t1[r]);
#pragma unroll
      for (int off = 1; off < 16; off <<= 1)
        v = fmaxf(v, __shfl_xor(v, off, 32));
      mnew[r] = fmaxf(mrow[r], v);
      alpha[r] = (mrow[r] == -INFINITY) ? 0.f : exp2f(mrow[r] - mnew[r]);
      mrow[r] = mnew[r];
    }
    float p0[8], p1[8];
#pragma unroll
    for (int r = 0; r < 8; ++r) {
      const float mm = (mnew[r] == -INFINITY) ? 0.f : mnew[r];
      p0[r] = exp2f(t0[r] - mm);
      p1[r] = exp2f(t1[r] - mm);
      float s = p0[r] + p1[r];
#pragma unroll
      for (int off = 1; off < 16; off <<= 1)
        s += __shfl_xor(s, off, 32);
      lrow[r] = lrow[r] * alpha[r] + s;
    }
#pragma unroll
    for (int f = 0; f < 4; ++f)
#pragma unroll
      for (int r = 0; r < 8; ++r) o[f][r] *= alpha[r];

    // ---- P: C layout -> LDS -> A layout (per-wave, ordered via dscnt)
    {
      const int prh = (lane >> 4) * 8, pc = lane & 15;
#pragma unroll
      for (int r = 0; r < 8; ++r) {
        ldsp[(r + prh) * 32 + pc]      = f2bf(p0[r]);
        ldsp[(r + prh) * 32 + pc + 16] = f2bf(p1[r]);
      }
    }
    asm volatile("s_wait_dscnt 0" ::: "memory");
    Frag16 ap;
    {
      const int row = lane & 15, hi2 = (lane >> 4) * 16;
#pragma unroll
      for (int r = 0; r < 8; ++r) {
        const int kk = 2 * r + hi2;
        ap.u[r] = *(const unsigned*)&ldsp[row * 32 + kk];
      }
    }

    // ---- PV: B frags from V^T (dh-major => contiguous key pairs)
#pragma unroll
    for (int f = 0; f < 4; ++f) {
      Frag16 bv;
      load_b(bv, vpan + (long)(f * 16) * L_ + kb, L_, lane);
      o[f] = wmma_bf16(ap, bv, o[f]);
    }
  }

  // ---- normalize + store context (bf16, (B*L) x D)
  const int col = lane & 15, rowhi = (lane >> 4) * 8;
#pragma unroll
  for (int f = 0; f < 4; ++f) {
#pragma unroll
    for (int r = 0; r < 8; ++r) {
      const int q = qt * 16 + r + rowhi;
      const float val = o[f][r] / lrow[r];
      Ctx[((long)b * L_ + q) * D_ + h * DH_ + f * 16 + col] = f2bf(val);
    }
  }
}

// ---------------- stage 3: output projection GEMM (f32 out + bias) ----------------
__global__ __launch_bounds__(256) void gemm_out_kernel(
    const unsigned short* __restrict__ A,    // M_ x D_ bf16 (ctx)
    const unsigned short* __restrict__ Bt,   // D_ x D_ bf16 (w_out^T)
    const float* __restrict__ bias,          // D_
    float* __restrict__ out) {               // M_ x D_ f32
  __shared__ unsigned short ldsB[2][64 * 32];
  const int tid = threadIdx.x, lane = tid & 31, wave = tid >> 5;
  const int n0 = blockIdx.x * 64;
  const int m0 = blockIdx.y * 128 + wave * 16;

  v8f c[4];
#pragma unroll
  for (int f = 0; f < 4; ++f)
#pragma unroll
    for (int i = 0; i < 8; ++i) c[f][i] = 0.f;

  gemm_mainloop(A + (long)m0 * D_, Bt + (long)n0 * D_, &ldsB[0][0], tid, lane, c);

  const int col = lane & 15, rowhi = (lane >> 4) * 8;
#pragma unroll
  for (int f = 0; f < 4; ++f) {
    const int n = n0 + f * 16 + col;
    const float bv = bias[n];
#pragma unroll
    for (int r = 0; r < 8; ++r) {
      const int m = m0 + r + rowhi;
      out[(long)m * D_ + n] = c[f][r] + bv;
    }
  }
}

// ---------------- launcher ----------------
extern "C" void kernel_launch(void* const* d_in, const int* in_sizes, int n_in,
                              void* d_out, int out_size, void* d_ws, size_t ws_size,
                              hipStream_t stream) {
  const float* x      = (const float*)d_in[0];
  const float* w_qkv  = (const float*)d_in[1];
  const float* b_qkv  = (const float*)d_in[2];
  const float* w_out  = (const float*)d_in[3];
  const float* b_out  = (const float*)d_in[4];
  const unsigned char* mask = (const unsigned char*)d_in[5];
  float* out = (float*)d_out;

  char* ws = (char*)d_ws;
  unsigned short* x_bf  = (unsigned short*)ws; ws += (size_t)M_ * D_ * 2;      // 8 MB
  unsigned short* wqkvT = (unsigned short*)ws; ws += (size_t)N1_ * D_ * 2;     // 6 MB
  unsigned short* woutT = (unsigned short*)ws; ws += (size_t)D_ * D_ * 2;      // 2 MB
  unsigned short* Qd    = (unsigned short*)ws; ws += (size_t)B_ * H_ * L_ * DH_ * 2;
  unsigned short* Kd    = (unsigned short*)ws; ws += (size_t)B_ * H_ * L_ * DH_ * 2;
  unsigned short* Vt    = (unsigned short*)ws; ws += (size_t)B_ * H_ * L_ * DH_ * 2;
  unsigned short* Ctx   = (unsigned short*)ws; ws += (size_t)M_ * D_ * 2;      // 8 MB

  cvt_bf16_kernel<<<(M_ * D_ + 255) / 256, 256, 0, stream>>>(x, x_bf, M_ * D_);
  cvt_t_bf16_kernel<<<(D_ * N1_ + 255) / 256, 256, 0, stream>>>(w_qkv, wqkvT, D_, N1_);
  cvt_t_bf16_kernel<<<(D_ * D_ + 255) / 256, 256, 0, stream>>>(w_out, woutT, D_, D_);

  gemm_qkv_kernel<<<dim3(N1_ / 64, M_ / 128), 256, 0, stream>>>(x_bf, wqkvT, b_qkv, Qd, Kd, Vt);
  attn_kernel<<<(B_ * H_ * (L_ / 16)) / 4, 128, 0, stream>>>(Qd, Kd, Vt, mask, Ctx);
  gemm_out_kernel<<<dim3(D_ / 64, M_ / 128), 256, 0, stream>>>(Ctx, woutT, b_out, out);
}
